// Attention_73452530696451
// MI455X (gfx1250) — matrix operand order repaired and checked by reading, not compiled, not run on hardware
//
#include <hip/hip_runtime.h>

// ---------------------------------------------------------------------------
// MI455X (gfx1250) GQA attention pipeline, bf16 WMMA everywhere.
//   B=2, T=2048, C=4096, H=32, KVH=8, HS=128
// Round 2: block-cooperative K/V staging through LDS via the Tensor Data
// Mover (tensor_load_to_lds + s_wait_tensorcnt), killing the 8x duplicated
// per-wave K/V global loads of round 1.
// ---------------------------------------------------------------------------

#define BDIM 2
#define TDIM 2048
#define CDIM 4096
#define HDIM 32
#define KVHDIM 8
#define HSDIM 128

typedef __attribute__((ext_vector_type(16))) __bf16 v16bf;
typedef __attribute__((ext_vector_type(8)))  float  v8f;
typedef unsigned int  u32x4 __attribute__((ext_vector_type(4)));
typedef int           i32x8 __attribute__((ext_vector_type(8)));
typedef int           i32x4 __attribute__((ext_vector_type(4)));

union FragBF {
    v16bf v;
    unsigned int   u[8];
    unsigned short s[16];
};

__device__ __forceinline__ unsigned short f2bf_u(float f) {
    unsigned int u = __float_as_uint(f);
    u += 0x7fffu + ((u >> 16) & 1u);          // round-to-nearest-even
    return (unsigned short)(u >> 16);
}
__device__ __forceinline__ unsigned int pack2bf(float a, float b) {
    return (unsigned int)f2bf_u(a) | ((unsigned int)f2bf_u(b) << 16);
}

#if __has_builtin(__builtin_amdgcn_tensor_load_to_lds)
#define HAVE_TDM 1
#else
#define HAVE_TDM 0
#endif

#if HAVE_TDM
// 2D tile DMA: global (row-major, 2-byte elems, row stride 1024 elems)
// -> LDS tile [rows=32][cols=128], zero-fill for rows >= rem_rows.
// D# layout per CDNA5 ISA.
__device__ __forceinline__ void tdm_load_tile_2d(unsigned lds_addr,
                                                 const unsigned short* gsrc,
                                                 int rem_rows)
{
    unsigned long long ga = (unsigned long long)(size_t)gsrc;
    u32x4 g0;
    g0[0] = 1u;                                   // count=1, user descriptor
    g0[1] = lds_addr;                             // lds_addr (bytes)
    g0[2] = (unsigned)ga;                         // global_addr[31:0]
    g0[3] = (unsigned)((ga >> 32) & 0x1FFFFFFu)   // global_addr[56:32]
          | (2u << 30);                           // type=2 ("image")
    i32x8 g1;
    g1[0] = (int)(1u << 16);                      // data_size=1 (2 bytes)
    g1[1] = (int)(128u << 16);                    // tensor_dim0 = 128
    g1[2] = (int)(((unsigned)rem_rows & 0xffffu) << 16); // tensor_dim1 lo16
    g1[3] = (int)(128u << 16);                    // tensor_dim1 hi16=0 | tile_dim0=128
    g1[4] = 32;                                   // tile_dim1=32, tile_dim2=0
    g1[5] = (int)(KVHDIM * HSDIM);                // tensor_dim0_stride = 1024
    g1[6] = 0;                                    // stride hi | dim1_stride lo
    g1[7] = 0;
    i32x4 gz = {};
#if __clang_major__ >= 23
    i32x8 gz8 = {};
    __builtin_amdgcn_tensor_load_to_lds(g0, g1, gz, gz, gz8, 0);
#else
    __builtin_amdgcn_tensor_load_to_lds(g0, g1, gz, gz, 0);
#endif
}
#endif

// ---------------------------------------------------------------------------
// Kernel 1: generic tiled GEMM  C[M,N] = A[M,K] @ B[K,N] + bias
//   fp32 inputs converted to bf16 on the LDS-staging path,
//   v_wmma_f32_16x16x32_bf16 compute, fp32 or bf16 output.
//   Block: 256 threads = 8 waves (4 M-slices x 2 N-slices), wave tile 32x64.
// ---------------------------------------------------------------------------
#define GBM 128
#define GBN 128
#define GBK 32

__global__ __launch_bounds__(256)
void gemm_bf16_wmma(const float* __restrict__ A, const float* __restrict__ Bw,
                    const float* __restrict__ bias, void* __restrict__ Cout,
                    int M, int N, int K, int outBF16)
{
    __shared__ unsigned short As[GBM][GBK + 2];   // [row][k]
    __shared__ unsigned short Bs[GBN][GBK + 2];   // transposed: [n][k]

    const int tid   = threadIdx.x;
    const int lane  = tid & 31;
    const int w     = tid >> 5;
    const int p     = lane >> 4;     // half-wave (0/1)
    const int nl    = lane & 15;     // lane within half
    const int warpM = w & 3;
    const int warpN = w >> 2;
    const int bm0   = blockIdx.y * GBM;
    const int bn0   = blockIdx.x * GBN;

    v8f acc[2][4];
    #pragma unroll
    for (int i = 0; i < 2; ++i)
        #pragma unroll
        for (int j = 0; j < 4; ++j) acc[i][j] = (v8f){};

    for (int k0 = 0; k0 < K; k0 += GBK) {
        // ---- stage A tile (128 x 32 fp32 -> bf16) -------------------------
        #pragma unroll
        for (int i = 0; i < 4; ++i) {
            int lin = i * 256 + tid;            // 0..1023 float4 slots
            int row = lin >> 3;
            int kq  = (lin & 7) << 2;
            const float4 f = *(const float4*)(A + (size_t)(bm0 + row) * K + k0 + kq);
            *(unsigned int*)&As[row][kq]     = pack2bf(f.x, f.y);
            *(unsigned int*)&As[row][kq + 2] = pack2bf(f.z, f.w);
        }
        // ---- stage B tile (32 x 128 fp32 -> bf16, transposed) -------------
        #pragma unroll
        for (int i = 0; i < 4; ++i) {
            int lin = i * 256 + tid;
            int kk  = lin >> 5;
            int n4  = (lin & 31) << 2;
            const float4 f = *(const float4*)(Bw + (size_t)(k0 + kk) * N + bn0 + n4);
            Bs[n4 + 0][kk] = f2bf_u(f.x);
            Bs[n4 + 1][kk] = f2bf_u(f.y);
            Bs[n4 + 2][kk] = f2bf_u(f.z);
            Bs[n4 + 3][kk] = f2bf_u(f.w);
        }
        if (k0 + GBK < K) {   // hint next tiles into cache: global_prefetch_b8
            __builtin_prefetch(A + (size_t)(bm0 + (tid >> 1)) * K + k0 + GBK, 0, 0);
            __builtin_prefetch(Bw + (size_t)(k0 + GBK + (tid >> 3)) * N + bn0, 0, 0);
        }
        __syncthreads();

        // ---- fragments (A 16x32 / B 32x16, contiguous k = 16*p + s in both) -------------
        FragBF af[2], bfv[4];
        #pragma unroll
        for (int am = 0; am < 2; ++am) {
            const int row = warpM * 32 + am * 16 + nl;
            #pragma unroll
            for (int v = 0; v < 8; ++v) {
                const int kk = p * 16 + 2 * v;
                af[am].u[v] = *(const unsigned int*)&As[row][kk];
            }
        }
        #pragma unroll
        for (int bn = 0; bn < 4; ++bn) {
            const int col = warpN * 64 + bn * 16 + nl;
            #pragma unroll
            for (int v = 0; v < 8; ++v)
                bfv[bn].u[v] = *(const unsigned int*)&Bs[col][p * 16 + 2 * v];
        }
        #pragma unroll
        for (int am = 0; am < 2; ++am)
            #pragma unroll
            for (int bn = 0; bn < 4; ++bn)
                acc[am][bn] = __builtin_amdgcn_wmma_f32_16x16x32_bf16(
                    false, af[am].v, false, bfv[bn].v,
                    (short)0, acc[am][bn], false, false);
        __syncthreads();
    }

    // ---- epilogue: bias add, fp32 or bf16 store ---------------------------
    #pragma unroll
    for (int am = 0; am < 2; ++am) {
        #pragma unroll
        for (int bn = 0; bn < 4; ++bn) {
            const int gcol = bn0 + warpN * 64 + bn * 16 + nl;
            const float bv = bias[gcol];
            #pragma unroll
            for (int e = 0; e < 8; ++e) {
                const int grow = bm0 + warpM * 32 + am * 16 + p * 8 + e;
                const float val = acc[am][bn][e] + bv;
                if (outBF16)
                    ((unsigned short*)Cout)[(size_t)grow * N + gcol] = f2bf_u(val);
                else
                    ((float*)Cout)[(size_t)grow * N + gcol] = val;
            }
        }
    }
}

// ---------------------------------------------------------------------------
// Kernel 2: RoPE, fp32 in -> bf16 out.  One thread per (row, head, d<64) pair.
// ---------------------------------------------------------------------------
__global__ __launch_bounds__(256)
void rope_to_bf16(const float* __restrict__ x, unsigned short* __restrict__ out,
                  int heads)
{
    const int idx   = blockIdx.x * blockDim.x + threadIdx.x;
    const int d     = idx & 63;
    const int h     = (idx >> 6) % heads;
    const int row   = idx / (64 * heads);     // row in [0, B*T)
    const int t     = row & (TDIM - 1);
    const int HHS   = heads * HSDIM;

    const float inv = __expf(-(float)d * (9.210340371976184f / 64.0f)); // 10000^(-d/64)
    const float ang = (float)t * inv;
    const float c   = __cosf(ang);
    const float s   = __sinf(ang);

    const size_t base = (size_t)row * HHS + h * HSDIM + d;
    const float x0 = x[base];
    const float x1 = x[base + 64];
    out[base]      = f2bf_u(x0 * c - x1 * s);
    out[base + 64] = f2bf_u(x1 * c + x0 * s);
}

// ---------------------------------------------------------------------------
// Kernel 3: causal flash attention with WMMA.
//   Block = 256 threads = 8 waves; each wave owns 16 query rows.
//   Grid  = (T/128, H, B).  Per 32-key block:
//     - K/V tile (32x128 bf16 each) staged ONCE per block into LDS by the
//       Tensor Data Mover (wave 0 issues tensor_load_to_lds, waits
//       s_wait_tensorcnt 0, then the workgroup barrier publishes it),
//     - 8 WMMAs for S = Q K^T (dword LDS fragment reads),
//     - online softmax in the WMMA C-layout (shfl reductions per 16-lane half),
//     - P re-laid out through per-wave LDS (s_wait_dscnt ordering),
//     - 8 WMMAs for O += P V.
// ---------------------------------------------------------------------------
__global__ __launch_bounds__(256)
void flash_attn_wmma(const unsigned short* __restrict__ qb16,
                     const unsigned short* __restrict__ kb16,
                     const unsigned short* __restrict__ vb16,
                     float* __restrict__ yout)
{
    __shared__ __align__(16) unsigned short Ks[32][HSDIM];  // [key][d]
    __shared__ __align__(16) unsigned short Vs[32][HSDIM];  // [key][d]
    __shared__ float Plds[8][16][33];   // per-wave P tile re-layout buffer

    const int tid  = threadIdx.x;
    const int lane = tid & 31;
    const int w    = tid >> 5;
    const int p    = lane >> 4;
    const int nl   = lane & 15;

    const int b   = blockIdx.z;
    const int h   = blockIdx.y;
    const int kvh = h >> 2;                         // H/KVH = 4
    const int qt  = blockIdx.x;
    const int qrow0 = qt * 128 + w * 16;

    const unsigned short* qp = qb16 + (size_t)b * TDIM * (HDIM * HSDIM) + h * HSDIM;
    const unsigned short* kp = kb16 + (size_t)b * TDIM * (KVHDIM * HSDIM) + kvh * HSDIM;
    const unsigned short* vp = vb16 + (size_t)b * TDIM * (KVHDIM * HSDIM) + kvh * HSDIM;

    // ---- Q fragments: 4 chunks of 16x32 bf16 (A-layout), dword loads ------
    FragBF qf[4];
    #pragma unroll
    for (int c = 0; c < 4; ++c)
        #pragma unroll
        for (int v = 0; v < 8; ++v) {
            const int kk = c * 32 + p * 16 + 2 * v;
            qf[c].u[v] = *(const unsigned int*)(qp + (size_t)(qrow0 + nl) * (HDIM * HSDIM) + kk);
        }

    v8f acc[8];
    #pragma unroll
    for (int dt = 0; dt < 8; ++dt) acc[dt] = (v8f){};
    float mrow[8], lrow[8];
    #pragma unroll
    for (int e = 0; e < 8; ++e) { mrow[e] = -1e30f; lrow[e] = 0.0f; }

    const float scale = 0.08838834764831845f;       // 1/sqrt(128)
    const int kmax = qt * 128 + 127;                // block-uniform causal bound

    for (int key0 = 0; key0 <= kmax; key0 += 32) {
        __syncthreads();   // previous tile fully consumed before overwrite
#if HAVE_TDM
        if (w == 0) {
            tdm_load_tile_2d((unsigned)(size_t)&Ks[0][0],
                             kp + (size_t)key0 * (KVHDIM * HSDIM), TDIM - key0);
            tdm_load_tile_2d((unsigned)(size_t)&Vs[0][0],
                             vp + (size_t)key0 * (KVHDIM * HSDIM), TDIM - key0);
#if __has_builtin(__builtin_amdgcn_s_wait_tensorcnt)
            __builtin_amdgcn_s_wait_tensorcnt(0);
#else
            asm volatile("s_wait_tensorcnt 0x0" ::: "memory");
#endif
        }
#else
        #pragma unroll
        for (int i = 0; i < 2; ++i) {
            const int lin = i * 256 + tid;
            const int key = lin >> 4;
            const int d8  = (lin & 15) << 3;
            const int gk  = min(key0 + key, TDIM - 1);
            *(uint4*)&Ks[key][d8] = *(const uint4*)(kp + (size_t)gk * (KVHDIM * HSDIM) + d8);
            *(uint4*)&Vs[key][d8] = *(const uint4*)(vp + (size_t)gk * (KVHDIM * HSDIM) + d8);
        }
#endif
        __syncthreads();   // publish K/V tile

        if (key0 > qrow0 + 15) continue;   // fully-masked tile for this wave

        // ---- S = Q @ K^T for two 16-key sub-tiles -------------------------
        float S[2][8];
        #pragma unroll
        for (int s = 0; s < 2; ++s) {
            v8f sa = (v8f){};
            #pragma unroll
            for (int c = 0; c < 4; ++c) {
                FragBF bf;
                #pragma unroll
                for (int v = 0; v < 8; ++v)
                    bf.u[v] = *(const unsigned int*)&Ks[s * 16 + nl][c * 32 + p * 16 + 2 * v];
                sa = __builtin_amdgcn_wmma_f32_16x16x32_bf16(
                        false, qf[c].v, false, bf.v, (short)0, sa, false, false);
            }
            #pragma unroll
            for (int e = 0; e < 8; ++e) {
                const int qr = qrow0 + p * 8 + e;
                const int kc = key0 + s * 16 + nl;
                S[s][e] = (kc > qr) ? -1e30f : sa[e] * scale;
            }
        }
        // ---- online softmax update ---------------------------------------
        float alpha[8];
        #pragma unroll
        for (int e = 0; e < 8; ++e) {
            float cand = fmaxf(S[0][e], S[1][e]);
            #pragma unroll
            for (int off = 8; off >= 1; off >>= 1)     // reduce within 16-lane half
                cand = fmaxf(cand, __shfl_xor(cand, off, 32));
            const float mnew = fmaxf(mrow[e], cand);
            alpha[e] = __expf(mrow[e] - mnew);
            const float p0 = __expf(S[0][e] - mnew);
            const float p1 = __expf(S[1][e] - mnew);
            Plds[w][p * 8 + e][nl]      = p0;
            Plds[w][p * 8 + e][16 + nl] = p1;
            float rs = p0 + p1;
            #pragma unroll
            for (int off = 8; off >= 1; off >>= 1)
                rs += __shfl_xor(rs, off, 32);
            lrow[e] = lrow[e] * alpha[e] + rs;
            mrow[e] = mnew;
        }
        #pragma unroll
        for (int dt = 0; dt < 8; ++dt)
            #pragma unroll
            for (int e = 0; e < 8; ++e) acc[dt][e] *= alpha[e];

        // same-wave LDS ordering before C-layout -> A-layout reload
        asm volatile("s_wait_dscnt 0x0" ::: "memory");

        // ---- P fragment (A-layout 16x32) ----------------------------------
        FragBF pf;
        #pragma unroll
        for (int v = 0; v < 8; ++v) {
            const int kk = p * 16 + 2 * v;
            pf.u[v] = pack2bf(Plds[w][nl][kk], Plds[w][nl][kk + 1]);
        }
        // ---- O += P @ V ----------------------------------------------------
        #pragma unroll
        for (int dt = 0; dt < 8; ++dt) {
            FragBF vf;
            #pragma unroll
            for (int v = 0; v < 8; ++v) {
                const int d = dt * 16 + nl;
                vf.s[2 * v]     = Vs[p * 16 + 2 * v][d];
                vf.s[2 * v + 1] = Vs[p * 16 + 2 * v + 1][d];
            }
            acc[dt] = __builtin_amdgcn_wmma_f32_16x16x32_bf16(
                          false, pf.v, false, vf.v, (short)0, acc[dt], false, false);
        }
    }

    // ---- normalize + store y (fp32, [B*T, H*HS]) --------------------------
    float invl[8];
    #pragma unroll
    for (int e = 0; e < 8; ++e) invl[e] = 1.0f / lrow[e];
    #pragma unroll
    for (int dt = 0; dt < 8; ++dt)
        #pragma unroll
        for (int e = 0; e < 8; ++e) {
            const int qr = qrow0 + p * 8 + e;
            yout[((size_t)b * TDIM + qr) * (HDIM * HSDIM) + h * HSDIM + dt * 16 + nl]
                = acc[dt][e] * invl[e];
        }
}

// ---------------------------------------------------------------------------
// Host launcher
// ---------------------------------------------------------------------------
extern "C" void kernel_launch(void* const* d_in, const int* in_sizes, int n_in,
                              void* d_out, int out_size, void* d_ws, size_t ws_size,
                              hipStream_t stream)
{
    const float* q_x  = (const float*)d_in[0];
    const float* kv_x = (const float*)d_in[1];
    const float* Wq   = (const float*)d_in[2];
    const float* bq   = (const float*)d_in[3];
    const float* Wk   = (const float*)d_in[4];
    const float* bk   = (const float*)d_in[5];
    const float* Wv   = (const float*)d_in[6];
    const float* bv   = (const float*)d_in[7];
    const float* Wo   = (const float*)d_in[8];
    const float* bo   = (const float*)d_in[9];

    const int M   = BDIM * TDIM;          // 4096
    const int NQ  = HDIM * HSDIM;         // 4096
    const int NKV = KVHDIM * HSDIM;       // 1024

    char* ws = (char*)d_ws;
    float*          q_f32 = (float*)(ws);                                   // 64 MB
    float*          k_f32 = (float*)(ws + (size_t)67108864);                // 16 MB
    unsigned short* q_bf  = (unsigned short*)(ws + (size_t)83886080);       // 32 MB
    unsigned short* k_bf  = (unsigned short*)(ws + (size_t)117440512);      //  8 MB
    unsigned short* v_bf  = (unsigned short*)(ws + (size_t)125829120);      //  8 MB
    float*          y_f32 = (float*)(ws + (size_t)134217728);               // 64 MB

    // QKV projections (bf16 WMMA GEMMs)
    gemm_bf16_wmma<<<dim3(NQ  / GBN, M / GBM), 256, 0, stream>>>(q_x,  Wq, bq, q_f32, M, NQ,  CDIM, 0);
    gemm_bf16_wmma<<<dim3(NKV / GBN, M / GBM), 256, 0, stream>>>(kv_x, Wk, bk, k_f32, M, NKV, CDIM, 0);
    gemm_bf16_wmma<<<dim3(NKV / GBN, M / GBM), 256, 0, stream>>>(kv_x, Wv, bv, v_bf,  M, NKV, CDIM, 1);

    // RoPE + downconvert to bf16
    rope_to_bf16<<<(M * HDIM   * 64) / 256, 256, 0, stream>>>(q_f32, q_bf, HDIM);
    rope_to_bf16<<<(M * KVHDIM * 64) / 256, 256, 0, stream>>>(k_f32, k_bf, KVHDIM);

    // Causal flash attention (TDM-staged K/V tiles)
    flash_attn_wmma<<<dim3(TDIM / 128, HDIM, BDIM), 256, 0, stream>>>(q_bf, k_bf, v_bf, y_f32);

    // Output projection -> d_out (fp32)
    gemm_bf16_wmma<<<dim3(CDIM / GBN, M / GBM), 256, 0, stream>>>(y_f32, Wo, bo, (float*)d_out, M, CDIM, NQ, 0);
}
